// SelfAttention_24627342475583
// MI455X (gfx1250) — matrix operand order repaired and checked by reading, not compiled, not run on hardware
//
#include <hip/hip_runtime.h>
#include <hip/hip_bf16.h>
#include <math.h>
#include <stdint.h>

typedef __bf16 bf16;
typedef __bf16 v16bf __attribute__((ext_vector_type(16)));
typedef __bf16 v8bf  __attribute__((ext_vector_type(8)));
typedef __bf16 v4bf  __attribute__((ext_vector_type(4)));
typedef float  v8f   __attribute__((ext_vector_type(8)));

__device__ __forceinline__ v8f wmma_bf16(v16bf a, v16bf b, v8f c) {
    // D(16x16 f32) = A(16x32 bf16) x B(32x16 bf16) + C
    return __builtin_amdgcn_wmma_f32_16x16x32_bf16(
        false, a, false, b, (short)0, c, false, false);
}

// Build a v16bf A/B fragment from two 16-byte (8 x bf16) aligned loads.
__device__ __forceinline__ v16bf ld2(const bf16* p0, const bf16* p1) {
    union { v16bf v; v8bf h[2]; } u;
    u.h[0] = *(const v8bf*)p0;
    u.h[1] = *(const v8bf*)p1;
    return u.v;
}

// Async global -> LDS 16-byte copy (CDNA5 GLOBAL_LOAD_ASYNC_TO_LDS_B128,
// tracked by ASYNCcnt).  Low 32 bits of a generic shared pointer are the
// LDS byte address (flat->LDS aperture truncation).
__device__ __forceinline__ void async_cp16(void* lds, const void* g) {
    const uint32_t l = (uint32_t)(uintptr_t)lds;
    const uint64_t a = (uint64_t)(uintptr_t)g;
    asm volatile("global_load_async_to_lds_b128 %0, %1, off"
                 :: "v"(l), "v"(a) : "memory");
}
__device__ __forceinline__ void wait_async0() {
    asm volatile("s_wait_asynccnt 0" ::: "memory");
}

// ---------------------------------------------------------------------------
// Split f32 -> (hi, lo) bf16 pair.  x ~= hi + lo with ~16 mantissa bits.
// ---------------------------------------------------------------------------
__global__ __launch_bounds__(256)
void split_kernel(const float* __restrict__ X, bf16* __restrict__ H,
                  bf16* __restrict__ L) {
    const size_t i = (size_t)blockIdx.x * 256 + threadIdx.x;  // 4 elems each
    const float4 v = *(const float4*)(X + i * 4);
    const float xs[4] = {v.x, v.y, v.z, v.w};
    v4bf h, l;
#pragma unroll
    for (int u = 0; u < 4; ++u) {
        const bf16 hb = (bf16)xs[u];
        h[u] = hb;
        l[u] = (bf16)(xs[u] - (float)hb);
    }
    *(v4bf*)(H + i * 4) = h;
    *(v4bf*)(L + i * 4) = l;
}

// ---------------------------------------------------------------------------
// GEMM: C[M,N] = A[M,K] @ W[N,K]^T, split-bf16 inputs, f32 accum.
// Block 256 thr (8 waves), tile 128(M) x 256(N); wave tile 64x64 (4x4 WMMA
// accums); K staged 32 wide via async global->LDS copies.
// vtrans=1 writes bf16 output transposed per head: [b][h][d][s] (for V).
// ---------------------------------------------------------------------------
#define GST 40  // bf16 LDS row stride (mult of 8 -> 16B-aligned b128 frags)

__global__ __launch_bounds__(256)
void gemm_split_kernel(const bf16* __restrict__ Ah, const bf16* __restrict__ Al,
                       const bf16* __restrict__ Wh, const bf16* __restrict__ Wl,
                       float* __restrict__ Cf, bf16* __restrict__ Ch,
                       bf16* __restrict__ Cl, int M, int N, int K, int vtrans) {
    __shared__ __align__(16) bf16 sAh[128 * GST];
    __shared__ __align__(16) bf16 sAl[128 * GST];
    __shared__ __align__(16) bf16 sWh[256 * GST];
    __shared__ __align__(16) bf16 sWl[256 * GST];

    const int tid  = threadIdx.x;
    const int wave = tid >> 5;
    const int lane = tid & 31;
    const int wm   = wave >> 2;        // 0..1
    const int wn   = wave & 3;         // 0..3
    const int lm   = lane & 15;
    const int g    = lane >> 4;

    const int m0 = blockIdx.y * 128;
    const int n0 = blockIdx.x * 256;

    const int sr = tid >> 1;           // staging row 0..127
    const int sc = (tid & 1) * 16;     // staging col 0 or 16

    v8f acc[4][4];
#pragma unroll
    for (int i = 0; i < 4; ++i)
#pragma unroll
        for (int j = 0; j < 4; ++j) acc[i][j] = (v8f){};

    for (int k0 = 0; k0 < K; k0 += 32) {
        __syncthreads();  // all waves done reading previous stage
        {
            const size_t ga = (size_t)(m0 + sr) * K + k0 + sc;
            async_cp16(&sAh[sr * GST + sc],     Ah + ga);
            async_cp16(&sAh[sr * GST + sc + 8], Ah + ga + 8);
            async_cp16(&sAl[sr * GST + sc],     Al + ga);
            async_cp16(&sAl[sr * GST + sc + 8], Al + ga + 8);
#pragma unroll
            for (int rr = 0; rr < 2; ++rr) {
                const size_t gw = (size_t)(n0 + rr * 128 + sr) * K + k0 + sc;
                const int lr = (rr * 128 + sr) * GST + sc;
                async_cp16(&sWh[lr],     Wh + gw);
                async_cp16(&sWh[lr + 8], Wh + gw + 8);
                async_cp16(&sWl[lr],     Wl + gw);
                async_cp16(&sWl[lr + 8], Wl + gw + 8);
            }
        }
        wait_async0();
        __syncthreads();

        // B fragments once per stage (reused 4x each)
        v16bf b_h[4], b_l[4];
#pragma unroll
        for (int j = 0; j < 4; ++j) {
            const int row = wn * 64 + j * 16 + lm;
            b_h[j] = ld2(&sWh[row * GST + 8 * g], &sWh[row * GST + 16 + 8 * g]);
            b_l[j] = ld2(&sWl[row * GST + 8 * g], &sWl[row * GST + 16 + 8 * g]);
        }
#pragma unroll
        for (int i = 0; i < 4; ++i) {
            const int row = wm * 64 + i * 16 + lm;
            const v16bf a_h = ld2(&sAh[row * GST + 8 * g], &sAh[row * GST + 16 + 8 * g]);
            const v16bf a_l = ld2(&sAl[row * GST + 8 * g], &sAl[row * GST + 16 + 8 * g]);
#pragma unroll
            for (int j = 0; j < 4; ++j) {
                acc[i][j] = wmma_bf16(a_h, b_h[j], acc[i][j]);
                acc[i][j] = wmma_bf16(a_h, b_l[j], acc[i][j]);
                acc[i][j] = wmma_bf16(a_l, b_h[j], acc[i][j]);
            }
        }
    }

    const int rowadd = g << 3;
    const int rbase  = m0 + wm * 64;
    const int cbase  = n0 + wn * 64;
#pragma unroll
    for (int i = 0; i < 4; ++i)
#pragma unroll
        for (int j = 0; j < 4; ++j)
#pragma unroll
            for (int e = 0; e < 8; ++e) {
                const int row = rbase + i * 16 + e + rowadd;
                const int col = cbase + j * 16 + lm;
                const float v = acc[i][j][e];
                if (Cf) Cf[(size_t)row * N + col] = v;
                if (Ch) {
                    const bf16 hb = (bf16)v;
                    const bf16 lb = (bf16)(v - (float)hb);
                    size_t idx;
                    if (vtrans) {
                        // [b][h][d][s]: b=row>>11, s=row&2047, h=col>>7, d=col&127
                        idx = (((size_t)(row >> 11) * 16 + (col >> 7)) * 128 +
                               (col & 127)) * 2048 + (row & 2047);
                    } else {
                        idx = (size_t)row * N + col;
                    }
                    Ch[idx] = hb;
                    Cl[idx] = lb;
                }
            }
}

// ---------------------------------------------------------------------------
// Causal flash attention on split-bf16 Q/K (row-major) and V (pre-transposed
// [b][h][d][s]); f32 softmax; bf16 hi/lo output.
// Grid (S/64, H, B); 128 threads = 4 waves; wave owns 16 query rows.
// K/V^T tiles staged via async global->LDS copies.
// ---------------------------------------------------------------------------
#define KT 136  // Kt [key][d] bf16 stride
#define VT 40   // Vt [d][key] bf16 stride
#define PT 40   // P  [row][key] bf16 stride

__global__ __launch_bounds__(128)
void flash_attn_kernel(const bf16* __restrict__ Qh_, const bf16* __restrict__ Ql_,
                       const bf16* __restrict__ Kh_, const bf16* __restrict__ Kl_,
                       const bf16* __restrict__ VTh_, const bf16* __restrict__ VTl_,
                       bf16* __restrict__ Oh_, bf16* __restrict__ Ol_,
                       int S, int D) {
    __shared__ __align__(16) bf16 Kth[32 * KT];
    __shared__ __align__(16) bf16 Ktl[32 * KT];
    __shared__ __align__(16) bf16 Vth[128 * VT];
    __shared__ __align__(16) bf16 Vtl[128 * VT];
    __shared__ __align__(16) bf16 Psh[4 * 16 * PT];
    __shared__ __align__(16) bf16 Psl[4 * 16 * PT];

    const int tid  = threadIdx.x;
    const int wave = tid >> 5;
    const int lane = tid & 31;
    const int lm   = lane & 15;
    const int g    = lane >> 4;
    const int rowadd = g << 3;

    const int b  = blockIdx.z;
    const int h  = blockIdx.y;
    const int q0 = blockIdx.x * 64;
    const int qw0 = q0 + wave * 16;

    const size_t headoff = (size_t)b * S * D + (size_t)h * 128;
    const bf16* Qhh = Qh_ + headoff;
    const bf16* Qll = Ql_ + headoff;
    const bf16* Khh = Kh_ + headoff;
    const bf16* Kll = Kl_ + headoff;
    const size_t headoffV = ((size_t)b * 16 + h) * 128 * S;  // [b][h][d][s]
    const bf16* Vth_g = VTh_ + headoffV;
    const bf16* Vtl_g = VTl_ + headoffV;

    // Q fragments in registers: 4 K-chunks x (hi,lo)
    v16bf qh[4], ql[4];
    {
        const bf16* qr0 = Qhh + (size_t)(qw0 + lm) * D;
        const bf16* qr1 = Qll + (size_t)(qw0 + lm) * D;
#pragma unroll
        for (int c = 0; c < 4; ++c) {
            qh[c] = ld2(qr0 + 32 * c + 8 * g, qr0 + 32 * c + 16 + 8 * g);
            ql[c] = ld2(qr1 + 32 * c + 8 * g, qr1 + 32 * c + 16 + 8 * g);
        }
    }

    v8f o[8];
#pragma unroll
    for (int t = 0; t < 8; ++t) o[t] = (v8f){};
    float mrow[8], lrow[8];
#pragma unroll
    for (int e = 0; e < 8; ++e) { mrow[e] = -INFINITY; lrow[e] = 0.0f; }

    const float sscale = 0.08838834764831845f;  // 1/sqrt(128)
    const float LOG2E  = 1.44269504088896340f;
    bf16* Pwh = &Psh[wave * 16 * PT];
    bf16* Pwl = &Psl[wave * 16 * PT];

    const int jmax = (q0 + 63) / 32;

    for (int j = 0; j <= jmax; ++j) {
        const int kv0 = j * 32;
        __syncthreads();
        // Stage K [key][d] via async copies
#pragma unroll
        for (int it = 0; it < 4; ++it) {
            const int idx  = tid + it * 128;    // 0..511
            const int key  = idx >> 4;          // 0..31
            const int dseg = (idx & 15) * 8;    // 0,8,..,120
            const size_t gsrc = (size_t)(kv0 + key) * D + dseg;
            async_cp16(&Kth[key * KT + dseg], Khh + gsrc);
            async_cp16(&Ktl[key * KT + dseg], Kll + gsrc);
        }
        // Stage V^T [d][key] via async copies from pre-transposed global
#pragma unroll
        for (int it = 0; it < 4; ++it) {
            const int idx = tid + it * 128;     // 0..511
            const int d   = idx >> 2;           // 0..127
            const int ks  = (idx & 3) * 8;      // 0,8,16,24
            const size_t gsrc = (size_t)d * S + kv0 + ks;
            async_cp16(&Vth[d * VT + ks], Vth_g + gsrc);
            async_cp16(&Vtl[d * VT + ks], Vtl_g + gsrc);
        }
        wait_async0();
        __syncthreads();

        if (kv0 <= qw0 + 15) {  // wave-uniform causal guard
            // S = (Q @ K^T) * sscale : 3-term split bf16
            v8f st[2];
            st[0] = (v8f){}; st[1] = (v8f){};
#pragma unroll
            for (int c = 0; c < 4; ++c) {
#pragma unroll
                for (int t = 0; t < 2; ++t) {
                    const bf16* kp0 = &Kth[(t * 16 + lm) * KT + 32 * c + 8 * g];
                    const bf16* kp1 = &Ktl[(t * 16 + lm) * KT + 32 * c + 8 * g];
                    const v16bf bh = ld2(kp0, kp0 + 16);
                    const v16bf bl = ld2(kp1, kp1 + 16);
                    st[t] = wmma_bf16(qh[c], bh, st[t]);
                    st[t] = wmma_bf16(qh[c], bl, st[t]);
                    st[t] = wmma_bf16(ql[c], bh, st[t]);
                }
            }
            // scale + causal mask
#pragma unroll
            for (int t = 0; t < 2; ++t)
#pragma unroll
                for (int e = 0; e < 8; ++e) {
                    const int qrow = qw0 + e + rowadd;
                    const int kcol = kv0 + t * 16 + lm;
                    const float s = st[t][e] * sscale;
                    st[t][e] = (kcol <= qrow) ? s : -INFINITY;
                }
            // row max across 16-lane half-groups
            float rmax[8];
#pragma unroll
            for (int e = 0; e < 8; ++e) {
                float v = fmaxf(st[0][e], st[1][e]);
#pragma unroll
                for (int m = 1; m < 16; m <<= 1)
                    v = fmaxf(v, __shfl_xor(v, m, 32));
                rmax[e] = v;
            }
            // online softmax; write split-bf16 P to per-wave LDS
            float aLocal[8];
#pragma unroll
            for (int e = 0; e < 8; ++e) {
                const float mnew  = fmaxf(mrow[e], rmax[e]);
                const float alpha = exp2f((mrow[e] - mnew) * LOG2E);
                const float p0 = exp2f((st[0][e] - mnew) * LOG2E);
                const float p1 = exp2f((st[1][e] - mnew) * LOG2E);
                const bf16 p0h = (bf16)p0;
                const bf16 p1h = (bf16)p1;
                const int prow = (e + rowadd) * PT;
                Pwh[prow + lm]      = p0h;
                Pwh[prow + 16 + lm] = p1h;
                Pwl[prow + lm]      = (bf16)(p0 - (float)p0h);
                Pwl[prow + 16 + lm] = (bf16)(p1 - (float)p1h);
                float ps = p0 + p1;
#pragma unroll
                for (int m = 1; m < 16; m <<= 1)
                    ps += __shfl_xor(ps, m, 32);
                lrow[e] = lrow[e] * alpha + ps;
                mrow[e] = mnew;
                aLocal[e] = alpha;
            }
            // O = O*alpha + P @ V (3-term split)
#pragma unroll
            for (int t = 0; t < 8; ++t)
#pragma unroll
                for (int e = 0; e < 8; ++e)
                    o[t][e] = o[t][e] * aLocal[e];

            const v16bf ph = ld2(&Pwh[lm * PT + 8 * g], &Pwh[lm * PT + 16 + 8 * g]);
            const v16bf pl = ld2(&Pwl[lm * PT + 8 * g], &Pwl[lm * PT + 16 + 8 * g]);
#pragma unroll
            for (int t = 0; t < 8; ++t) {
                const bf16* vp0 = &Vth[(t * 16 + lm) * VT + 8 * g];
                const bf16* vp1 = &Vtl[(t * 16 + lm) * VT + 8 * g];
                const v16bf bh = ld2(vp0, vp0 + 16);
                const v16bf bl = ld2(vp1, vp1 + 16);
                o[t] = wmma_bf16(ph, bh, o[t]);
                o[t] = wmma_bf16(ph, bl, o[t]);
                o[t] = wmma_bf16(pl, bh, o[t]);
            }
        }
    }

    // normalize and write hi/lo bf16 output
#pragma unroll
    for (int t = 0; t < 8; ++t)
#pragma unroll
        for (int e = 0; e < 8; ++e) {
            const float val = o[t][e] / lrow[e];
            const size_t idx = headoff + (size_t)(qw0 + e + rowadd) * D + t * 16 + lm;
            const bf16 hb = (bf16)val;
            Oh_[idx] = hb;
            Ol_[idx] = (bf16)(val - (float)hb);
        }
}

// ---------------------------------------------------------------------------
extern "C" void kernel_launch(void* const* d_in, const int* in_sizes, int n_in,
                              void* d_out, int out_size, void* d_ws, size_t ws_size,
                              hipStream_t stream) {
    constexpr int Bb = 4, S = 2048, D = 2048, H = 16;
    constexpr int M = Bb * S;                        // 8192
    constexpr size_t eA = (size_t)M * D;             // activations
    constexpr size_t eW = (size_t)D * D;             // weights

    const float* x  = (const float*)d_in[0];
    const float* wq = (const float*)d_in[1];
    const float* wk = (const float*)d_in[2];
    const float* wv = (const float*)d_in[3];
    const float* wo = (const float*)d_in[4];
    float* out = (float*)d_out;

    bf16* p = (bf16*)d_ws;
    bf16* xh = p;  p += eA;  bf16* xl = p;  p += eA;
    bf16* wqh = p; p += eW;  bf16* wql = p; p += eW;
    bf16* wkh = p; p += eW;  bf16* wkl = p; p += eW;
    bf16* wvh = p; p += eW;  bf16* wvl = p; p += eW;
    bf16* woh = p; p += eW;  bf16* wol = p; p += eW;
    bf16* Qh = p;  p += eA;  bf16* Ql = p;  p += eA;
    bf16* Kh = p;  p += eA;  bf16* Kl = p;  p += eA;
    bf16* VTh = p; p += eA;  bf16* VTl = p; p += eA;   // transposed [b][h][d][s]
    bf16* Aoh = p; p += eA;  bf16* Aol = p; p += eA;

    // 1) split f32 -> bf16 hi/lo
    split_kernel<<<dim3(eA / 1024), dim3(256), 0, stream>>>(x, xh, xl);
    split_kernel<<<dim3(eW / 1024), dim3(256), 0, stream>>>(wq, wqh, wql);
    split_kernel<<<dim3(eW / 1024), dim3(256), 0, stream>>>(wk, wkh, wkl);
    split_kernel<<<dim3(eW / 1024), dim3(256), 0, stream>>>(wv, wvh, wvl);
    split_kernel<<<dim3(eW / 1024), dim3(256), 0, stream>>>(wo, woh, wol);

    dim3 gemmGrid(D / 256, M / 128);  // (8, 64)
    // 2) Q/K/V projections (V written transposed per head)
    gemm_split_kernel<<<gemmGrid, dim3(256), 0, stream>>>(
        xh, xl, wqh, wql, nullptr, Qh, Ql, M, D, D, 0);
    gemm_split_kernel<<<gemmGrid, dim3(256), 0, stream>>>(
        xh, xl, wkh, wkl, nullptr, Kh, Kl, M, D, D, 0);
    gemm_split_kernel<<<gemmGrid, dim3(256), 0, stream>>>(
        xh, xl, wvh, wvl, nullptr, VTh, VTl, M, D, D, 1);

    // 3) causal flash attention
    flash_attn_kernel<<<dim3(S / 64, H, Bb), dim3(128), 0, stream>>>(
        Qh, Ql, Kh, Kl, VTh, VTl, Aoh, Aol, S, D);

    // 4) output projection -> f32 result
    gemm_split_kernel<<<gemmGrid, dim3(256), 0, stream>>>(
        Aoh, Aol, woh, wol, out, nullptr, nullptr, M, D, D, 0);
}
